// RGATLayer_59107339927615
// MI455X (gfx1250) — matrix operand order repaired and checked by reading, not compiled, not run on hardware
//
#include <hip/hip_runtime.h>
#include <cstdint>

// ---------------- common types / helpers ----------------
typedef __attribute__((ext_vector_type(16))) __bf16 v16bf;
typedef __attribute__((ext_vector_type(8)))  float  v8f;
typedef unsigned int u32;
typedef __attribute__((ext_vector_type(4)))  u32    u32x4;

union Frag16 {
    v16bf v;
    unsigned short u[16];
    u32x4 q[2];
};

__device__ __forceinline__ unsigned short f2bf(float f) {
    unsigned u = __float_as_uint(f);
    u += 0x7FFFu + ((u >> 16) & 1u);   // round to nearest even
    return (unsigned short)(u >> 16);
}

__device__ __forceinline__ v8f zero8() {
    v8f z = {0.f, 0.f, 0.f, 0.f, 0.f, 0.f, 0.f, 0.f};
    return z;
}

__device__ __forceinline__ v8f wmma_bf16(const Frag16& a, const Frag16& b, v8f c) {
    return __builtin_amdgcn_wmma_f32_16x16x32_bf16(false, a.v, false, b.v,
                                                   (short)0, c, false, false);
}

// CDNA5 async global->LDS copy (16B per lane), tracked by ASYNCcnt.
__device__ __forceinline__ void async_ld16(const unsigned short* g, unsigned short* l) {
    unsigned lds = (unsigned)(size_t)l;
    unsigned long long ga = (unsigned long long)(size_t)g;
    asm volatile("global_load_async_to_lds_b128 %0, %1, off"
                 :: "v"(lds), "v"(ga) : "memory");
}
__device__ __forceinline__ void async_wait0() {
    asm volatile("s_wait_asynccnt 0x0" ::: "memory");
}

// ---------------- elementwise converters ----------------
__global__ void k_f2b(const float* __restrict__ src, unsigned short* __restrict__ dst, int n) {
    int i = blockIdx.x * 256 + threadIdx.x;
    if (i < n) dst[i] = f2bf(src[i]);
}

// src [R][C] f32 -> dst [C][R] bf16  (i.e. B -> B^T)
__global__ void k_f2bT(const float* __restrict__ src, unsigned short* __restrict__ dst,
                       int R_, int C_) {
    int i = blockIdx.x * 256 + threadIdx.x;
    if (i < R_ * C_) {
        int r = i / C_, c = i % C_;
        dst[(size_t)c * R_ + r] = f2bf(src[i]);
    }
}

#define LSTR 40   // LDS row stride (elems): 16B-aligned rows, staggers banks

// ---------------- small/batched tiled WMMA GEMM (64x64 macro tile) ----------------
// C[M,N](f32 or bf16) = A[M,K](bf16,row-major) * Bt[N,K]^T (bf16, B^T row-major) [+bias][relu]
template <int BIAS, int RELU, int OBF>
__global__ __launch_bounds__(128) void k_gemm(
    const unsigned short* __restrict__ A, long long sA,
    const unsigned short* __restrict__ Bt, long long sB,
    const float* __restrict__ bias,
    void* __restrict__ Cv, long long sC,
    int M, int N, int K)
{
    __shared__ unsigned short As[64 * LSTR];
    __shared__ unsigned short Bs[64 * LSTR];

    const int t = threadIdx.x;
    const int wave = t >> 5, lane = t & 31;
    const int half = lane >> 4, l16 = lane & 15;
    const int m0 = blockIdx.y * 64;
    const int n0 = blockIdx.x * 64;
    const int z  = blockIdx.z;
    A  += (size_t)z * sA;
    Bt += (size_t)z * sB;

    const int wm = (wave >> 1) * 32;
    const int wn = (wave & 1) * 32;

    v8f acc[2][2];
    for (int a = 0; a < 2; a++)
        for (int b = 0; b < 2; b++) acc[a][b] = zero8();

    const int lr = t >> 1;          // 0..63
    const int lc = (t & 1) * 16;    // 0 or 16

    for (int k0 = 0; k0 < K; k0 += 32) {
        if (m0 + lr < M) {
            const unsigned short* src = A + (size_t)(m0 + lr) * K + k0 + lc;
            *(u32x4*)&As[lr * LSTR + lc]     = ((const u32x4*)src)[0];
            *(u32x4*)&As[lr * LSTR + lc + 8] = ((const u32x4*)src)[1];
        } else {
            u32x4 zq = {0u, 0u, 0u, 0u};
            *(u32x4*)&As[lr * LSTR + lc] = zq;
            *(u32x4*)&As[lr * LSTR + lc + 8] = zq;
        }
        if (n0 + lr < N) {
            const unsigned short* src = Bt + (size_t)(n0 + lr) * K + k0 + lc;
            *(u32x4*)&Bs[lr * LSTR + lc]     = ((const u32x4*)src)[0];
            *(u32x4*)&Bs[lr * LSTR + lc + 8] = ((const u32x4*)src)[1];
        } else {
            u32x4 zq = {0u, 0u, 0u, 0u};
            *(u32x4*)&Bs[lr * LSTR + lc] = zq;
            *(u32x4*)&Bs[lr * LSTR + lc + 8] = zq;
        }
        __syncthreads();

        Frag16 af[2], bf_[2];
        for (int a = 0; a < 2; a++) {
            const unsigned short* p = &As[(wm + a * 16 + l16) * LSTR];
            af[a].q[0] = *(const u32x4*)(p + 8 * half);
            af[a].q[1] = *(const u32x4*)(p + 16 + 8 * half);
        }
        for (int b = 0; b < 2; b++) {
            const unsigned short* p = &Bs[(wn + b * 16 + l16) * LSTR];
            bf_[b].q[0] = *(const u32x4*)(p + 8 * half);
            bf_[b].q[1] = *(const u32x4*)(p + 16 + 8 * half);
        }
        for (int a = 0; a < 2; a++)
            for (int b = 0; b < 2; b++)
                acc[a][b] = wmma_bf16(af[a], bf_[b], acc[a][b]);
        __syncthreads();
    }

    for (int a = 0; a < 2; a++)
        for (int b = 0; b < 2; b++) {
            int col = n0 + wn + b * 16 + l16;
            float bv = BIAS ? bias[col] : 0.0f;
            for (int r = 0; r < 8; r++) {
                int row = m0 + wm + a * 16 + r + 8 * half;
                if (row < M && col < N) {
                    float v = acc[a][b][r] + bv;
                    if (RELU) v = v > 0.f ? v : 0.f;
                    if (OBF)
                        ((unsigned short*)Cv)[(size_t)z * sC + (size_t)row * N + col] = f2bf(v);
                    else
                        ((float*)Cv)[(size_t)z * sC + (size_t)row * N + col] = v;
                }
            }
        }
}

// ---------------- big WMMA GEMM: 128x128 macro tile, async-to-LDS double buffering ----
// Requires M%128==0, N%128==0, K%32==0. 256 threads = 8 waves (4M x 2N), wave tile 32x64.
template <int BIAS, int RELU, int OBF>
__global__ __launch_bounds__(256) void k_gemm_big(
    const unsigned short* __restrict__ A,
    const unsigned short* __restrict__ Bt,
    const float* __restrict__ bias,
    void* __restrict__ Cv,
    int M, int N, int K)
{
    __shared__ unsigned short As[2][128 * LSTR];
    __shared__ unsigned short Bs[2][128 * LSTR];

    const int t = threadIdx.x;
    const int wave = t >> 5, lane = t & 31;
    const int half = lane >> 4, l16 = lane & 15;
    const int m0 = blockIdx.y * 128;
    const int n0 = blockIdx.x * 128;

    const int wm = (wave >> 1) * 32;   // 0,32,64,96
    const int wn = (wave & 1) * 64;    // 0,64

    v8f acc[2][4];
    for (int a = 0; a < 2; a++)
        for (int b = 0; b < 4; b++) acc[a][b] = zero8();

    const int lr = t >> 1;             // 0..127
    const int lc = (t & 1) * 16;       // 0 or 16
    const unsigned short* srcA = A + (size_t)(m0 + lr) * K + lc;
    const unsigned short* srcB = Bt + (size_t)(n0 + lr) * K + lc;

    // stage tile 0 into buffer 0 via CDNA5 async global->LDS
    async_ld16(srcA,     &As[0][lr * LSTR + lc]);
    async_ld16(srcA + 8, &As[0][lr * LSTR + lc + 8]);
    async_ld16(srcB,     &Bs[0][lr * LSTR + lc]);
    async_ld16(srcB + 8, &Bs[0][lr * LSTR + lc + 8]);
    async_wait0();
    __syncthreads();

    int buf = 0;
    for (int k0 = 0; k0 < K; k0 += 32) {
        const int nxt = buf ^ 1;
        const bool more = (k0 + 32) < K;
        if (more) {
            const unsigned short* pA = srcA + k0 + 32;
            const unsigned short* pB = srcB + k0 + 32;
            async_ld16(pA,     &As[nxt][lr * LSTR + lc]);
            async_ld16(pA + 8, &As[nxt][lr * LSTR + lc + 8]);
            async_ld16(pB,     &Bs[nxt][lr * LSTR + lc]);
            async_ld16(pB + 8, &Bs[nxt][lr * LSTR + lc + 8]);
        }

        Frag16 af[2], bf_[4];
        for (int a = 0; a < 2; a++) {
            const unsigned short* p = &As[buf][(wm + a * 16 + l16) * LSTR];
            af[a].q[0] = *(const u32x4*)(p + 8 * half);
            af[a].q[1] = *(const u32x4*)(p + 16 + 8 * half);
        }
        for (int b = 0; b < 4; b++) {
            const unsigned short* p = &Bs[buf][(wn + b * 16 + l16) * LSTR];
            bf_[b].q[0] = *(const u32x4*)(p + 8 * half);
            bf_[b].q[1] = *(const u32x4*)(p + 16 + 8 * half);
        }
        for (int a = 0; a < 2; a++)
            for (int b = 0; b < 4; b++)
                acc[a][b] = wmma_bf16(af[a], bf_[b], acc[a][b]);

        if (more) async_wait0();
        __syncthreads();
        buf = nxt;
    }

    for (int a = 0; a < 2; a++)
        for (int b = 0; b < 4; b++) {
            int col = n0 + wn + b * 16 + l16;
            float bv = BIAS ? bias[col] : 0.0f;
            for (int r = 0; r < 8; r++) {
                int row = m0 + wm + a * 16 + r + 8 * half;
                float v = acc[a][b][r] + bv;
                if (RELU) v = v > 0.f ? v : 0.f;
                if (OBF)
                    ((unsigned short*)Cv)[(size_t)row * N + col] = f2bf(v);
                else
                    ((float*)Cv)[(size_t)row * N + col] = v;
            }
        }
}

// ---------------- QKV split: f32 [2048,1536] -> Qbf/Kbf [bh,512,64], Vt [bh,64,512] ----------------
__global__ void k_split_qkv(const float* __restrict__ qkv,
                            unsigned short* __restrict__ Q,
                            unsigned short* __restrict__ K_,
                            unsigned short* __restrict__ Vt) {
    int i = blockIdx.x * 256 + threadIdx.x;
    if (i >= 2048 * 1536) return;
    int row = i / 1536, c = i % 1536;
    int b = row >> 9, tok = row & 511;
    int which = c >> 9, hid = c & 511, h = hid >> 6, d = hid & 63;
    unsigned short v = f2bf(qkv[i]);
    size_t bh = (size_t)(b * 8 + h);
    if (which == 0)       Q[(bh * 512 + tok) * 64 + d] = v;
    else if (which == 1)  K_[(bh * 512 + tok) * 64 + d] = v;
    else                  Vt[(bh * 64 + d) * 512 + tok] = v;
}

// ---------------- fused relational attention ----------------
// grid (32 i-tiles, 32 bh), block 128 (4 waves). One 16-row i-tile per workgroup.
__global__ __launch_bounds__(128) void k_attn(
    const unsigned short* __restrict__ Qb,   // [bh,512,64] bf16
    const unsigned short* __restrict__ Kb,   // [bh,512,64] bf16
    const unsigned short* __restrict__ Vt,   // [bh,64,512] bf16 (V^T)
    const float* __restrict__ qr,            // [bh,512,128] f32
    const int* __restrict__ rel,             // [b,512,512]
    const unsigned char* __restrict__ msk,   // [b,512,512]
    float* __restrict__ attn_out,            // [bh,512,64] f32
    unsigned short* __restrict__ wsum_bf)    // [bh,512,128] bf16
{
    const int bh = blockIdx.y;
    const int b = bh >> 3;
    const int i0 = blockIdx.x * 16;
    const int t = threadIdx.x, wave = t >> 5, lane = t & 31;
    const int half = lane >> 4, l16 = lane & 15;

    __shared__ float sS[16 * 512];
    __shared__ float sOut[16 * 64];
    __shared__ float sWsum[16 * 128];
    __shared__ float sRed[16 * 8];
    __shared__ float sMax[16];
    __shared__ float sInv[16];

    for (int idx = t; idx < 16 * 64; idx += 128) sOut[idx] = 0.f;
    for (int idx = t; idx < 16 * 128; idx += 128) sWsum[idx] = 0.f;

    // Q A-fragments direct from global (all waves share same i-tile)
    const unsigned short* qrow = Qb + ((size_t)bh * 512 + i0 + l16) * 64;
    Frag16 aLo, aHi;
    aLo.q[0] = *(const u32x4*)(qrow + 8 * half);
    aLo.q[1] = *(const u32x4*)(qrow + 16 + 8 * half);
    aHi.q[0] = *(const u32x4*)(qrow + 32 + 8 * half);
    aHi.q[1] = *(const u32x4*)(qrow + 48 + 8 * half);
    __syncthreads();

    // ---- scores: e = (Q.K + qr[rel]) / 8, mask -> -1e20 ----
    for (int jt = wave; jt < 32; jt += 4) {
        int j0 = jt * 16;
        const unsigned short* krow = Kb + ((size_t)bh * 512 + j0 + l16) * 64;
        Frag16 bLo, bHi;
        bLo.q[0] = *(const u32x4*)(krow + 8 * half);
        bLo.q[1] = *(const u32x4*)(krow + 16 + 8 * half);
        bHi.q[0] = *(const u32x4*)(krow + 32 + 8 * half);
        bHi.q[1] = *(const u32x4*)(krow + 48 + 8 * half);
        v8f c = zero8();
        c = wmma_bf16(aLo, bLo, c);
        c = wmma_bf16(aHi, bHi, c);
        int j = j0 + l16;
        for (int r = 0; r < 8; r++) {
            int m = r + 8 * half;
            int i = i0 + m;
            int rr = rel[((size_t)b * 512 + i) * 512 + j];
            float e = (c[r] + qr[((size_t)bh * 512 + i) * 128 + rr]) * 0.125f;
            if (msk[((size_t)b * 512 + i) * 512 + j]) e = -1e20f;
            sS[m * 512 + j] = e;
        }
    }
    __syncthreads();

    // ---- softmax: row max ----
    {
        int m = t >> 3, seg = t & 7;
        float mx = -3.0e38f;
        const float* row = &sS[m * 512 + seg * 64];
        for (int c = 0; c < 64; c++) mx = fmaxf(mx, row[c]);
        sRed[m * 8 + seg] = mx;
    }
    __syncthreads();
    if (t < 16) {
        float mx = sRed[t * 8];
        for (int s = 1; s < 8; s++) mx = fmaxf(mx, sRed[t * 8 + s]);
        sMax[t] = mx;
    }
    __syncthreads();
    // ---- exp + row sum (unnormalized p kept in sS) ----
    {
        int m = t >> 3, seg = t & 7;
        float mx = sMax[m], sum = 0.f;
        float* row = &sS[m * 512 + seg * 64];
        for (int c = 0; c < 64; c++) { float p = __expf(row[c] - mx); row[c] = p; sum += p; }
        sRed[m * 8 + seg] = sum;
    }
    __syncthreads();
    if (t < 16) {
        float s = 0.f;
        for (int q = 0; q < 8; q++) s += sRed[t * 8 + q];
        sInv[t] = 1.0f / s;
    }
    __syncthreads();

    // ---- per-relation binning of attention weights ----
    {
        int m = t >> 3, seg = t & 7;
        int i = i0 + m;
        const int* rrow = &rel[((size_t)b * 512 + i) * 512 + seg * 64];
        const float* prow = &sS[m * 512 + seg * 64];
        for (int c = 0; c < 64; c++) atomicAdd(&sWsum[m * 128 + rrow[c]], prow[c]);
    }

    // ---- out = P @ V  (K-dim = 32 contiguous j per wmma) ----
    v8f accD[4];
    for (int q = 0; q < 4; q++) accD[q] = zero8();
    for (int p = wave; p < 16; p += 4) {
        int jb = p * 32;
        Frag16 af;
        {
            const float* r0 = &sS[l16 * 512 + jb + 8 * half];
            const float* r1 = &sS[l16 * 512 + jb + 16 + 8 * half];
            for (int e = 0; e < 8; e++) af.u[e] = f2bf(r0[e]);
            for (int e = 0; e < 8; e++) af.u[8 + e] = f2bf(r1[e]);
        }
        for (int q = 0; q < 4; q++) {
            int d = q * 16 + l16;
            const unsigned short* vrow = Vt + ((size_t)bh * 64 + d) * 512 + jb;
            Frag16 bf_;
            bf_.q[0] = *(const u32x4*)(vrow + 8 * half);
            bf_.q[1] = *(const u32x4*)(vrow + 16 + 8 * half);
            accD[q] = wmma_bf16(af, bf_, accD[q]);
        }
    }
    for (int q = 0; q < 4; q++) {
        int d = q * 16 + l16;
        for (int r = 0; r < 8; r++) atomicAdd(&sOut[(r + 8 * half) * 64 + d], accD[q][r]);
    }
    __syncthreads();

    // ---- normalized writes ----
    for (int idx = t; idx < 16 * 64; idx += 128) {
        int m = idx >> 6, d = idx & 63;
        attn_out[((size_t)bh * 512 + i0 + m) * 64 + d] = sOut[idx] * sInv[m];
    }
    for (int idx = t; idx < 16 * 128; idx += 128) {
        int m = idx >> 7, rr = idx & 127;
        wsum_bf[((size_t)bh * 512 + i0 + m) * 128 + rr] = f2bf(sWsum[idx] * sInv[m]);
    }
}

// ---------------- merge heads: (attn + rel_out)[bh,512,64] -> bf16 [b*512, 512] ----------------
__global__ void k_merge(const float* __restrict__ attn, const float* __restrict__ rel_out,
                        unsigned short* __restrict__ xout) {
    int i = blockIdx.x * 256 + threadIdx.x;
    if (i >= 4 * 8 * 512 * 64) return;
    int d = i & 63, tok = (i >> 6) & 511, bh = i >> 15;
    int b = bh >> 3, h = bh & 7;
    float v = attn[i] + rel_out[i];
    xout[((size_t)(b * 512) + tok) * 512 + h * 64 + d] = f2bf(v);
}

// ---------------- residual + LayerNorm over 512 ----------------
__global__ __launch_bounds__(256) void k_resid_ln(
    const float* __restrict__ resid, const float* __restrict__ y,
    const float* __restrict__ g, const float* __restrict__ bb,
    float* __restrict__ outf, unsigned short* __restrict__ outb) {
    int row = blockIdx.x, t = threadIdx.x;
    __shared__ float s1[256], s2[256];
    float v0 = resid[(size_t)row * 512 + t] + y[(size_t)row * 512 + t];
    float v1 = resid[(size_t)row * 512 + 256 + t] + y[(size_t)row * 512 + 256 + t];
    s1[t] = v0 + v1;
    s2[t] = v0 * v0 + v1 * v1;
    __syncthreads();
    for (int o = 128; o > 0; o >>= 1) {
        if (t < o) { s1[t] += s1[t + o]; s2[t] += s2[t + o]; }
        __syncthreads();
    }
    float mean = s1[0] * (1.f / 512.f);
    float var = s2[0] * (1.f / 512.f) - mean * mean;
    float rstd = rsqrtf(var + 1e-5f);
    float o0 = (v0 - mean) * rstd * g[t] + bb[t];
    float o1 = (v1 - mean) * rstd * g[t + 256] + bb[t + 256];
    outf[(size_t)row * 512 + t] = o0;
    outf[(size_t)row * 512 + 256 + t] = o1;
    if (outb) {
        outb[(size_t)row * 512 + t] = f2bf(o0);
        outb[(size_t)row * 512 + 256 + t] = f2bf(o1);
    }
}

// ---------------- host launcher ----------------
static inline int cdiv(int a, int b) { return (a + b - 1) / b; }

extern "C" void kernel_launch(void* const* d_in, const int* in_sizes, int n_in,
                              void* d_out, int out_size, void* d_ws, size_t ws_size,
                              hipStream_t stream) {
    const float* q      = (const float*)d_in[0];
    const float* o      = (const float*)d_in[1];
    const int* qrel     = (const int*)d_in[2];
    const int* orel     = (const int*)d_in[3];
    const unsigned char* qmask = (const unsigned char*)d_in[4];
    const unsigned char* omask = (const unsigned char*)d_in[5];
    const float* qkv_w  = (const float*)d_in[6];
    const float* qkv_b  = (const float*)d_in[7];
    const float* ca_w   = (const float*)d_in[8];
    const float* ca_b   = (const float*)d_in[9];
    const float* ff_w1  = (const float*)d_in[10];
    const float* ff_b1  = (const float*)d_in[11];
    const float* ff_w2  = (const float*)d_in[12];
    const float* ff_b2  = (const float*)d_in[13];
    const float* ln1_g  = (const float*)d_in[14];
    const float* ln1_bv = (const float*)d_in[15];
    const float* ln2_g  = (const float*)d_in[16];
    const float* ln2_bv = (const float*)d_in[17];
    const float* relk   = (const float*)d_in[18];
    const float* relv   = (const float*)d_in[19];

    char* ws = (char*)d_ws;
    size_t off = 0;
    auto alloc = [&](size_t bytes) -> void* {
        void* p = ws + off;
        off += (bytes + 255) & ~(size_t)255;
        return p;
    };
    unsigned short* qkv_wT = (unsigned short*)alloc((size_t)512 * 1536 * 2);
    unsigned short* ca_wT  = (unsigned short*)alloc((size_t)512 * 512 * 2);
    unsigned short* ff1T   = (unsigned short*)alloc((size_t)512 * 2048 * 2);
    unsigned short* ff2T   = (unsigned short*)alloc((size_t)2048 * 512 * 2);
    unsigned short* relk_b = (unsigned short*)alloc((size_t)128 * 64 * 2);
    unsigned short* relvT  = (unsigned short*)alloc((size_t)128 * 64 * 2);
    unsigned short* x_bf    = (unsigned short*)alloc((size_t)2048 * 512 * 2);
    float*          qkv_f   = (float*)alloc((size_t)2048 * 1536 * 4);
    unsigned short* Qbf     = (unsigned short*)alloc((size_t)32 * 512 * 64 * 2);
    unsigned short* Kbf     = (unsigned short*)alloc((size_t)32 * 512 * 64 * 2);
    unsigned short* Vtb     = (unsigned short*)alloc((size_t)32 * 64 * 512 * 2);
    float*          qrbuf   = (float*)alloc((size_t)32 * 512 * 128 * 4);
    unsigned short* wsum_bf = (unsigned short*)alloc((size_t)32 * 512 * 128 * 2);
    float*          attn_o  = (float*)alloc((size_t)32 * 512 * 64 * 4);
    float*          relo    = (float*)alloc((size_t)32 * 512 * 64 * 4);
    unsigned short* xatt_bf = (unsigned short*)alloc((size_t)2048 * 512 * 2);
    float*          ca_o    = (float*)alloc((size_t)2048 * 512 * 4);
    float*          ln1_f   = (float*)alloc((size_t)2048 * 512 * 4);
    unsigned short* ln1_b16 = (unsigned short*)alloc((size_t)2048 * 512 * 2);
    unsigned short* h1_bf   = (unsigned short*)alloc((size_t)2048 * 2048 * 2);
    float*          ff2_o   = (float*)alloc((size_t)2048 * 512 * 4);
    (void)ws_size; (void)in_sizes; (void)n_in; (void)out_size;

    // weight conversion (B -> B^T bf16)
    k_f2bT<<<cdiv(512 * 1536, 256), 256, 0, stream>>>(qkv_w, qkv_wT, 512, 1536);
    k_f2bT<<<cdiv(512 * 512, 256), 256, 0, stream>>>(ca_w, ca_wT, 512, 512);
    k_f2bT<<<cdiv(512 * 2048, 256), 256, 0, stream>>>(ff_w1, ff1T, 512, 2048);
    k_f2bT<<<cdiv(2048 * 512, 256), 256, 0, stream>>>(ff_w2, ff2T, 2048, 512);
    k_f2b<<<cdiv(128 * 64, 256), 256, 0, stream>>>(relk, relk_b, 128 * 64);
    k_f2bT<<<cdiv(128 * 64, 256), 256, 0, stream>>>(relv, relvT, 128, 64);

    for (int s = 0; s < 2; s++) {
        const float* x = s ? o : q;
        const int* rel = s ? orel : qrel;
        const unsigned char* mk = s ? omask : qmask;
        float* outp = (float*)d_out + (size_t)s * 2048 * 512;

        k_f2b<<<cdiv(2048 * 512, 256), 256, 0, stream>>>(x, x_bf, 2048 * 512);
        // QKV projection (big tile + async staging)
        k_gemm_big<1, 0, 0><<<dim3(1536 / 128, 2048 / 128, 1), 256, 0, stream>>>(
            x_bf, qkv_wT, qkv_b, qkv_f, 2048, 1536, 512);
        k_split_qkv<<<cdiv(2048 * 1536, 256), 256, 0, stream>>>(qkv_f, Qbf, Kbf, Vtb);
        // qr[bh,i,r] = Q . rel_k_emb[r]   (batched over bh, B shared)
        k_gemm<0, 0, 0><<<dim3(128 / 64, 512 / 64, 32), 128, 0, stream>>>(
            Qbf, (long long)512 * 64, relk_b, 0, nullptr, qrbuf, (long long)512 * 128,
            512, 128, 64);
        // fused attention (scores + softmax + P@V + relation binning)
        k_attn<<<dim3(32, 32), 128, 0, stream>>>(Qbf, Kbf, Vtb, qrbuf, rel, mk,
                                                 attn_o, wsum_bf);
        // rel_out = wsum @ rel_v_emb  (batched over bh, B shared)
        k_gemm<0, 0, 0><<<dim3(64 / 64, 512 / 64, 32), 128, 0, stream>>>(
            wsum_bf, (long long)512 * 128, relvT, 0, nullptr, relo, (long long)512 * 64,
            512, 64, 128);
        k_merge<<<cdiv(32 * 512 * 64, 256), 256, 0, stream>>>(attn_o, relo, xatt_bf);
        // output projection
        k_gemm_big<1, 0, 0><<<dim3(512 / 128, 2048 / 128, 1), 256, 0, stream>>>(
            xatt_bf, ca_wT, ca_b, ca_o, 2048, 512, 512);
        k_resid_ln<<<2048, 256, 0, stream>>>(x, ca_o, ln1_g, ln1_bv, ln1_f, ln1_b16);
        // FFN
        k_gemm_big<1, 1, 1><<<dim3(2048 / 128, 2048 / 128, 1), 256, 0, stream>>>(
            ln1_b16, ff1T, ff_b1, h1_bf, 2048, 2048, 512);
        k_gemm_big<1, 0, 0><<<dim3(512 / 128, 2048 / 128, 1), 256, 0, stream>>>(
            h1_bf, ff2T, ff_b2, ff2_o, 2048, 512, 2048);
        k_resid_ln<<<2048, 256, 0, stream>>>(ln1_f, ff2_o, ln2_g, ln2_bv, outp, nullptr);
    }
}
